// SignEquivariantBlock_45844480917573
// MI455X (gfx1250) — compile-verified, matrix operand order checked by reading, not run on hardware
//
#include <hip/hip_runtime.h>
#include <hip/hip_bf16.h>
#include <hip/hip_fp16.h>

#define NN   20000      // nodes
#define NE   320000     // edges
#define FD   128        // feature dim
#define NH   8          // heads
#define EF   64         // edge feature dim
#define AD   128        // cond dim
#define NG   32         // groups
#define HID  512        // MLP hidden (4*F)
#define F2   256        // 2*F

typedef __attribute__((ext_vector_type(16))) _Float16 v16h;
typedef __attribute__((ext_vector_type(8)))  float    v8f;
typedef __attribute__((ext_vector_type(4)))  float    f32x4;

// ---------------- WMMA 16xN row-block helper ----------------
// A fragment (16x32 f16): lane l -> row m = l&15; per-lane K chunks at
// c0 = ((l>>4)<<3) and c0+16 (ISA 7.12.2 16-bit A layout). B mirrors it, so a
// K-major (transposed) weight matrix is loaded with the identical pattern.
// One wave computes a 16 x (NT*16) strip: each A fragment is loaded ONCE per
// k-step and reused against NT B fragments (weights are L2-resident), cutting
// activation re-reads NT-fold vs. one-tile-per-wave.
union frag16 { f32x4 q[2]; v16h h; };

template<int NT>
__device__ __forceinline__ void wmma_rowblock(const _Float16* __restrict__ A,
                                              const _Float16* __restrict__ BT,
                                              int arow, int bcol0, int K, int lane,
                                              v8f* acc) {
  const int chunk = (lane >> 4) << 3;
  const _Float16* ap  = A  + (size_t)arow * K + chunk;
  const _Float16* bp0 = BT + (size_t)(bcol0 + (lane & 15)) * K + chunk;
  for (int kb = 0; kb < K; kb += 32) {
    frag16 a;
    a.q[0] = *(const f32x4*)(ap + kb);
    a.q[1] = *(const f32x4*)(ap + kb + 16);
    #pragma unroll
    for (int t = 0; t < NT; ++t) {
      const _Float16* bp = bp0 + (size_t)t * 16 * K;
      frag16 b;
      b.q[0] = *(const f32x4*)(bp + kb);
      b.q[1] = *(const f32x4*)(bp + kb + 16);
      acc[t] = __builtin_amdgcn_wmma_f32_16x16x32_f16(false, a.h, false, b.h,
                                                      (short)0, acc[t], false, false);
    }
  }
}

template<int NT>
__device__ __forceinline__ void acc_zero(v8f* acc) {
  #pragma unroll
  for (int t = 0; t < NT; ++t)
    #pragma unroll
    for (int i = 0; i < 8; ++i) acc[t][i] = 0.f;
}

__device__ __forceinline__ float siluf(float x) { return x / (1.f + __expf(-x)); }

// monotone uint key for f32 atomic max
__device__ __forceinline__ unsigned fkey(float x) {
  unsigned u = __float_as_uint(x);
  return (u & 0x80000000u) ? ~u : (u | 0x80000000u);
}
__device__ __forceinline__ float funkey(unsigned k) {
  return __uint_as_float((k & 0x80000000u) ? (k & 0x7fffffffu) : ~k);
}

// ---------------- small prep kernels ----------------

__global__ __launch_bounds__(256) void k_sb(const float* __restrict__ cond,
    const float* __restrict__ Wmix, const float* __restrict__ Bmix,
    const float* __restrict__ Wmlp, const float* __restrict__ Bmlp,
    float* __restrict__ SBmix, float* __restrict__ SBmlp) {
  int idx = blockIdx.x * 256 + threadIdx.x;
  if (idx >= 2 * NG * F2) return;
  int j = idx % F2, g = (idx / F2) % NG, t = idx / (F2 * NG);
  const float* W = t ? Wmlp : Wmix;
  float sum = (t ? Bmlp : Bmix)[j];
  const float* c = cond + (size_t)g * AD;
  for (int k = 0; k < AD; ++k) sum += c[k] * W[(size_t)k * F2 + j];
  (t ? SBmlp : SBmix)[(size_t)g * F2 + j] = sum;
}

// transpose+convert weight [K,O] f32 -> [Opad,K] f16 (zero pad rows >= O)
__global__ __launch_bounds__(256) void k_wt(const float* __restrict__ W,
    _Float16* __restrict__ WT, int K, int O, int Opad) {
  int idx = blockIdx.x * 256 + threadIdx.x;
  if (idx >= Opad * K) return;
  int o = idx / K, k = idx - o * K;
  WT[idx] = (o < O) ? (_Float16)W[(size_t)k * O + o] : (_Float16)0.f;
}

// AdaLN over F=128 row, one wave per row -> f16
__global__ __launch_bounds__(256) void k_node_adaln(const float* __restrict__ X,
    const int* __restrict__ batch, const float* __restrict__ SB,
    _Float16* __restrict__ Y, int M) {
  int row  = blockIdx.x * 8 + (threadIdx.x >> 5);
  int lane = threadIdx.x & 31;
  if (row >= M) return;
  const float* x = X + (size_t)row * FD;
  float v[4], s = 0.f, sq = 0.f;
  #pragma unroll
  for (int j = 0; j < 4; ++j) { v[j] = x[lane + 32 * j]; s += v[j]; sq += v[j] * v[j]; }
  #pragma unroll
  for (int m = 16; m >= 1; m >>= 1) { s += __shfl_xor(s, m, 32); sq += __shfl_xor(sq, m, 32); }
  float mu  = s * (1.f / FD);
  float inv = rsqrtf(sq * (1.f / FD) - mu * mu + 1e-5f);
  const float* sb = SB + (size_t)batch[row] * F2;
  #pragma unroll
  for (int j = 0; j < 4; ++j) {
    int c = lane + 32 * j;
    Y[(size_t)row * FD + c] = (_Float16)((v[j] - mu) * inv * (1.f + sb[c]) + sb[FD + c]);
  }
}

// edge LN over EF=64 row, one wave per row: z = ln(a)*g + be -> f16
__global__ __launch_bounds__(256) void k_edge_ln(const float* __restrict__ A,
    const float* __restrict__ g, const float* __restrict__ be, _Float16* __restrict__ Z) {
  int row  = blockIdx.x * 8 + (threadIdx.x >> 5);
  int lane = threadIdx.x & 31;
  if (row >= NE) return;
  const float* x = A + (size_t)row * EF;
  float v0 = x[lane], v1 = x[lane + 32];
  float s = v0 + v1, sq = v0 * v0 + v1 * v1;
  #pragma unroll
  for (int m = 16; m >= 1; m >>= 1) { s += __shfl_xor(s, m, 32); sq += __shfl_xor(sq, m, 32); }
  float mu  = s * (1.f / EF);
  float inv = rsqrtf(sq * (1.f / EF) - mu * mu + 1e-5f);
  Z[(size_t)row * EF + lane]      = (_Float16)((v0 - mu) * inv * g[lane]      + be[lane]);
  Z[(size_t)row * EF + lane + 32] = (_Float16)((v1 - mu) * inv * g[lane + 32] + be[lane + 32]);
}

// ---------------- WMMA GEMM kernels (one wave = 16 x NT*16 strip) ----------------

// hh = y @ w_h + b_h   [N,128]x[128,256] -> f32 ; grid (N/16, 2), 128 cols/wave
__global__ __launch_bounds__(32) void k_gemm_hh(const _Float16* __restrict__ Y,
    const _Float16* __restrict__ WT, const float* __restrict__ bh, float* __restrict__ HH) {
  int lane = threadIdx.x, m0 = blockIdx.x * 16, c0 = blockIdx.y * 128;
  v8f acc[8]; acc_zero<8>(acc);
  wmma_rowblock<8>(Y, WT, m0 + (lane & 15), c0, FD, lane, acc);
  int rbase = m0 + ((lane >> 4) << 3);
  #pragma unroll
  for (int t = 0; t < 8; ++t) {
    int col = c0 + t * 16 + (lane & 15);
    float b = bh[col];
    #pragma unroll
    for (int r = 0; r < 8; ++r) HH[(size_t)(rbase + r) * F2 + col] = acc[t][r] + b;
  }
}

// m = silu(z@w_a + hsrc[src] + hdst[dst])  [E,64]x[64,128] -> f16 ; grid (E/16)
__global__ __launch_bounds__(32) void k_gemm_m(const _Float16* __restrict__ Z,
    const _Float16* __restrict__ WT, const float* __restrict__ HH,
    const int* __restrict__ edges, _Float16* __restrict__ M) {
  int lane = threadIdx.x, m0 = blockIdx.x * 16;
  v8f acc[8]; acc_zero<8>(acc);
  wmma_rowblock<8>(Z, WT, m0 + (lane & 15), 0, EF, lane, acc);
  int rbase = m0 + ((lane >> 4) << 3);
  #pragma unroll
  for (int r = 0; r < 8; ++r) {
    int e = rbase + r, s = edges[e], d = edges[NE + e];
    const float* hs = HH + (size_t)s * F2;
    const float* hd = HH + (size_t)d * F2 + FD;
    #pragma unroll
    for (int t = 0; t < 8; ++t) {
      int col = t * 16 + (lane & 15);
      float x = acc[t][r] + hs[col] + hd[col];
      M[(size_t)e * FD + col] = (_Float16)siluf(x);
    }
  }
}

// vw = m @ w_vatt  [E,128]x[128,144pad]: cols<128 -> v (f16), 128..135 -> logits (f32)
__global__ __launch_bounds__(32) void k_gemm_vw(const _Float16* __restrict__ M,
    const _Float16* __restrict__ WT, _Float16* __restrict__ V, float* __restrict__ WLOG) {
  int lane = threadIdx.x, m0 = blockIdx.x * 16;
  v8f acc[9]; acc_zero<9>(acc);
  wmma_rowblock<9>(M, WT, m0 + (lane & 15), 0, FD, lane, acc);
  int rbase = m0 + ((lane >> 4) << 3);
  #pragma unroll
  for (int t = 0; t < 8; ++t) {
    int col = t * 16 + (lane & 15);
    #pragma unroll
    for (int r = 0; r < 8; ++r)
      V[(size_t)(rbase + r) * FD + col] = (_Float16)acc[t][r];
  }
  if ((lane & 15) < NH) {
    int h = lane & 15;
    #pragma unroll
    for (int r = 0; r < 8; ++r)
      WLOG[(size_t)(rbase + r) * NH + h] = acc[8][r];
  }
}

// a_out = a + m @ w_edge + b_edge   [E,128]x[128,64] -> d_out (f32)
__global__ __launch_bounds__(32) void k_gemm_aout(const _Float16* __restrict__ M,
    const _Float16* __restrict__ WT, const float* __restrict__ Ain,
    const float* __restrict__ be, float* __restrict__ OutA) {
  int lane = threadIdx.x, m0 = blockIdx.x * 16;
  v8f acc[4]; acc_zero<4>(acc);
  wmma_rowblock<4>(M, WT, m0 + (lane & 15), 0, FD, lane, acc);
  int rbase = m0 + ((lane >> 4) << 3);
  #pragma unroll
  for (int t = 0; t < 4; ++t) {
    int col = t * 16 + (lane & 15);
    float b = be[col];
    #pragma unroll
    for (int r = 0; r < 8; ++r) {
      int e = rbase + r;
      OutA[(size_t)e * EF + col] = acc[t][r] + b + Ain[(size_t)e * EF + col];
    }
  }
}

// h1 = input + o @ w_out + b_out    [N,128]x[128,128] -> f32
__global__ __launch_bounds__(32) void k_gemm_h1(const _Float16* __restrict__ O16,
    const _Float16* __restrict__ WT, const float* __restrict__ X,
    const float* __restrict__ bo, float* __restrict__ H1) {
  int lane = threadIdx.x, m0 = blockIdx.x * 16;
  v8f acc[8]; acc_zero<8>(acc);
  wmma_rowblock<8>(O16, WT, m0 + (lane & 15), 0, FD, lane, acc);
  int rbase = m0 + ((lane >> 4) << 3);
  #pragma unroll
  for (int t = 0; t < 8; ++t) {
    int col = t * 16 + (lane & 15);
    float b = bo[col];
    #pragma unroll
    for (int r = 0; r < 8; ++r) {
      int n = rbase + r;
      H1[(size_t)n * FD + col] = acc[t][r] + b + X[(size_t)n * FD + col];
    }
  }
}

// u = silu(hn @ w1 + b1)   [N,128]x[128,512] -> f16 ; grid (N/16, 4)
__global__ __launch_bounds__(32) void k_gemm_u(const _Float16* __restrict__ HN,
    const _Float16* __restrict__ WT, const float* __restrict__ b1, _Float16* __restrict__ U) {
  int lane = threadIdx.x, m0 = blockIdx.x * 16, c0 = blockIdx.y * 128;
  v8f acc[8]; acc_zero<8>(acc);
  wmma_rowblock<8>(HN, WT, m0 + (lane & 15), c0, FD, lane, acc);
  int rbase = m0 + ((lane >> 4) << 3);
  #pragma unroll
  for (int t = 0; t < 8; ++t) {
    int col = c0 + t * 16 + (lane & 15);
    float b = b1[col];
    #pragma unroll
    for (int r = 0; r < 8; ++r)
      U[(size_t)(rbase + r) * HID + col] = (_Float16)siluf(acc[t][r] + b);
  }
}

// h_out = h1 + u @ w2 + b2   [N,512]x[512,128] -> d_out (f32)
__global__ __launch_bounds__(32) void k_gemm_hout(const _Float16* __restrict__ U,
    const _Float16* __restrict__ WT, const float* __restrict__ H1,
    const float* __restrict__ b2, float* __restrict__ OutH) {
  int lane = threadIdx.x, m0 = blockIdx.x * 16;
  v8f acc[8]; acc_zero<8>(acc);
  wmma_rowblock<8>(U, WT, m0 + (lane & 15), 0, HID, lane, acc);
  int rbase = m0 + ((lane >> 4) << 3);
  #pragma unroll
  for (int t = 0; t < 8; ++t) {
    int col = t * 16 + (lane & 15);
    float b = b2[col];
    #pragma unroll
    for (int r = 0; r < 8; ++r) {
      int n = rbase + r;
      OutH[(size_t)n * FD + col] = acc[t][r] + b + H1[(size_t)n * FD + col];
    }
  }
}

// ---------------- segmented softmax (f32 atomics, L2-resident) ----------------

__global__ __launch_bounds__(256) void k_init(unsigned* __restrict__ wmaxkey,
    float* __restrict__ wsum, float* __restrict__ o) {
  int idx = blockIdx.x * 256 + threadIdx.x;
  if (idx < NN * NH) { wmaxkey[idx] = fkey(-3.0e38f); wsum[idx] = 0.f; }
  if (idx < NN * FD) o[idx] = 0.f;
}

__global__ __launch_bounds__(256) void k_attmax(const float* __restrict__ WLOG,
    const int* __restrict__ edges, unsigned* __restrict__ wmaxkey) {
  int idx = blockIdx.x * 256 + threadIdx.x;
  if (idx >= NE * NH) return;
  int e = idx >> 3, h = idx & 7;
  atomicMax(&wmaxkey[(size_t)edges[e] * NH + h], fkey(WLOG[idx]));
}

__global__ __launch_bounds__(256) void k_attexp(const float* __restrict__ WLOG,
    const int* __restrict__ edges, const unsigned* __restrict__ wmaxkey,
    float* __restrict__ WEXP, float* __restrict__ wsum) {
  int idx = blockIdx.x * 256 + threadIdx.x;
  if (idx >= NE * NH) return;
  int e = idx >> 3, h = idx & 7, s = edges[e];
  float we = __expf(WLOG[idx] - funkey(wmaxkey[(size_t)s * NH + h]));
  WEXP[idx] = we;
  atomicAdd(&wsum[(size_t)s * NH + h], we);
}

__global__ __launch_bounds__(256) void k_attacc(const _Float16* __restrict__ V,
    const float* __restrict__ WEXP, const float* __restrict__ wsum,
    const int* __restrict__ edges, float* __restrict__ o) {
  int idx = blockIdx.x * 256 + threadIdx.x;
  if (idx >= NE * FD) return;
  int e = idx >> 7, c = idx & 127, h = c >> 4, s = edges[e];
  float coef = WEXP[(size_t)e * NH + h] / (wsum[(size_t)s * NH + h] + 1e-16f);
  atomicAdd(&o[(size_t)s * FD + c], coef * (float)V[idx]);
}

__global__ __launch_bounds__(256) void k_o2h(const float* __restrict__ o,
    _Float16* __restrict__ O16) {
  int idx = blockIdx.x * 256 + threadIdx.x;
  if (idx < NN * FD) O16[idx] = (_Float16)o[idx];
}

// ---------------- host launcher ----------------

extern "C" void kernel_launch(void* const* d_in, const int* in_sizes, int n_in,
                              void* d_out, int out_size, void* d_ws, size_t ws_size,
                              hipStream_t stream) {
  const float* input   = (const float*)d_in[0];
  const float* cond    = (const float*)d_in[1];
  const float* a_in    = (const float*)d_in[2];
  const int*   batch   = (const int*)  d_in[3];
  const int*   edges   = (const int*)  d_in[4];
  const float* w_amix  = (const float*)d_in[5];
  const float* b_amix  = (const float*)d_in[6];
  const float* w_h     = (const float*)d_in[7];
  const float* b_h     = (const float*)d_in[8];
  const float* w_a     = (const float*)d_in[9];
  const float* w_vatt  = (const float*)d_in[10];
  const float* w_out   = (const float*)d_in[11];
  const float* b_out   = (const float*)d_in[12];
  const float* g_edge  = (const float*)d_in[13];
  const float* be_edge = (const float*)d_in[14];
  const float* w_edge  = (const float*)d_in[15];
  const float* b_edge  = (const float*)d_in[16];
  const float* w_amlp  = (const float*)d_in[17];
  const float* b_amlp  = (const float*)d_in[18];
  const float* w1      = (const float*)d_in[19];
  const float* b1      = (const float*)d_in[20];
  const float* w2      = (const float*)d_in[21];
  const float* b2      = (const float*)d_in[22];
  (void)in_sizes; (void)n_in; (void)out_size; (void)ws_size;

  float* out_h = (float*)d_out;
  float* out_a = (float*)d_out + (size_t)NN * FD;

  char* ws = (char*)d_ws;
  size_t off = 0;
  auto carve = [&](size_t bytes) {
    char* p = ws + off; off = (off + bytes + 255) & ~(size_t)255; return p;
  };
  float*    sb_mix  = (float*)   carve((size_t)NG * F2 * 4);
  float*    sb_mlp  = (float*)   carve((size_t)NG * F2 * 4);
  _Float16* wT_h    = (_Float16*)carve((size_t)F2 * FD * 2);
  _Float16* wT_a    = (_Float16*)carve((size_t)FD * EF * 2);
  _Float16* wT_vatt = (_Float16*)carve((size_t)144 * FD * 2);  // O=136 padded to 144
  _Float16* wT_edge = (_Float16*)carve((size_t)EF * FD * 2);
  _Float16* wT_out  = (_Float16*)carve((size_t)FD * FD * 2);
  _Float16* wT_1    = (_Float16*)carve((size_t)HID * FD * 2);
  _Float16* wT_2    = (_Float16*)carve((size_t)FD * HID * 2);
  _Float16* y16     = (_Float16*)carve((size_t)NN * FD * 2);
  float*    hh      = (float*)   carve((size_t)NN * F2 * 4);
  _Float16* z16     = (_Float16*)carve((size_t)NE * EF * 2);
  _Float16* m16     = (_Float16*)carve((size_t)NE * FD * 2);
  _Float16* v16     = (_Float16*)carve((size_t)NE * FD * 2);
  float*    wlog    = (float*)   carve((size_t)NE * NH * 4);
  float*    wexp    = (float*)   carve((size_t)NE * NH * 4);
  unsigned* wmaxkey = (unsigned*)carve((size_t)NN * NH * 4);
  float*    wsum    = (float*)   carve((size_t)NN * NH * 4);
  float*    o_acc   = (float*)   carve((size_t)NN * FD * 4);
  _Float16* o16     = (_Float16*)carve((size_t)NN * FD * 2);
  float*    h1      = (float*)   carve((size_t)NN * FD * 4);
  _Float16* hn16    = (_Float16*)carve((size_t)NN * FD * 2);
  _Float16* u16     = (_Float16*)carve((size_t)NN * HID * 2);

  // 1) cond tables + weight transpose/convert to f16 (tiny, L2-resident)
  k_sb<<<(2 * NG * F2 + 255) / 256, 256, 0, stream>>>(cond, w_amix, b_amix, w_amlp, b_amlp, sb_mix, sb_mlp);
  k_wt<<<(F2 * FD + 255) / 256,  256, 0, stream>>>(w_h,    wT_h,    FD,  F2,  F2);
  k_wt<<<(FD * EF + 255) / 256,  256, 0, stream>>>(w_a,    wT_a,    EF,  FD,  FD);
  k_wt<<<(144 * FD + 255) / 256, 256, 0, stream>>>(w_vatt, wT_vatt, FD,  136, 144);
  k_wt<<<(EF * FD + 255) / 256,  256, 0, stream>>>(w_edge, wT_edge, FD,  EF,  EF);
  k_wt<<<(FD * FD + 255) / 256,  256, 0, stream>>>(w_out,  wT_out,  FD,  FD,  FD);
  k_wt<<<(HID * FD + 255) / 256, 256, 0, stream>>>(w1,     wT_1,    FD,  HID, HID);
  k_wt<<<(FD * HID + 255) / 256, 256, 0, stream>>>(w2,     wT_2,    HID, FD,  FD);

  // 2) node AdaLN(mix) -> y16; hh = y @ w_h + b_h
  k_node_adaln<<<NN / 8, 256, 0, stream>>>(input, batch, sb_mix, y16, NN);
  k_gemm_hh<<<dim3(NN / 16, 2), 32, 0, stream>>>(y16, wT_h, b_h, hh);

  // 3) edge LN -> z16; m = silu(z @ w_a + hsrc[src] + hdst[dst])
  k_edge_ln<<<NE / 8, 256, 0, stream>>>(a_in, g_edge, be_edge, z16);
  k_gemm_m<<<NE / 16, 32, 0, stream>>>(z16, wT_a, hh, edges, m16);

  // 4) vw = m @ w_vatt (values + logits); a_out = a + m @ w_edge + b_edge
  k_gemm_vw<<<NE / 16, 32, 0, stream>>>(m16, wT_vatt, v16, wlog);
  k_gemm_aout<<<NE / 16, 32, 0, stream>>>(m16, wT_edge, a_in, b_edge, out_a);

  // 5) segmented softmax + weighted scatter-sum into o
  k_init<<<(NN * FD + 255) / 256, 256, 0, stream>>>(wmaxkey, wsum, o_acc);
  k_attmax<<<(NE * NH + 255) / 256, 256, 0, stream>>>(wlog, edges, wmaxkey);
  k_attexp<<<(NE * NH + 255) / 256, 256, 0, stream>>>(wlog, edges, wmaxkey, wexp, wsum);
  k_attacc<<<(NE * FD + 255) / 256, 256, 0, stream>>>(v16, wexp, wsum, edges, o_acc);
  k_o2h<<<(NN * FD + 255) / 256, 256, 0, stream>>>(o_acc, o16);

  // 6) residual + MLP
  k_gemm_h1<<<NN / 16, 32, 0, stream>>>(o16, wT_out, input, b_out, h1);
  k_node_adaln<<<NN / 8, 256, 0, stream>>>(h1, batch, sb_mlp, hn16, NN);
  k_gemm_u<<<dim3(NN / 16, 4), 32, 0, stream>>>(hn16, wT_1, b1, u16);
  k_gemm_hout<<<NN / 16, 32, 0, stream>>>(u16, wT_2, h1, b2, out_h);
}
